// MRFLoss_42554535969045
// MI455X (gfx1250) — matrix operand order repaired and checked by reading, not compiled, not run on hardware
//
#include <hip/hip_runtime.h>

typedef __attribute__((ext_vector_type(16))) _Float16 v16h;
typedef __attribute__((ext_vector_type(8)))  _Float16 v8h;
typedef __attribute__((ext_vector_type(4)))  _Float16 v4h;
typedef __attribute__((ext_vector_type(8)))  float    v8f;

#define BATCH 2
#define TOPK  2
#define CCH   128
#define HW    48
#define NHW   46
#define NQ    (NHW*NHW)          /* 2116 queries per batch            */
#define NTOT  (TOPK*NQ)          /* 4232 candidates per batch         */
#define KDIM  (CCH*9)            /* 1152 = patch vector length        */
#define NSTEP (KDIM/32)          /* 36 K-steps of 32                  */
#define PPIX  (BATCH*HW*HW)      /* 4608 pred pixels                  */
#define TPIX  (BATCH*TOPK*HW*HW) /* 9216 target pixels                */

// ---------------------------------------------------------------------------
// gfx1250 async global->LDS copy (ASYNCcnt-tracked). VDST VGPR carries the
// wave-relative LDS byte address (low 32 bits of the generic LDS pointer).
// ---------------------------------------------------------------------------
__device__ __forceinline__ void async_cp16(unsigned lds_off, unsigned long long gaddr) {
    asm volatile("global_load_async_to_lds_b128 %0, %1, off"
                 :: "v"(lds_off), "v"(gaddr) : "memory");
}
__device__ __forceinline__ void wait_async0() {
#if __has_builtin(__builtin_amdgcn_s_wait_asynccnt)
    __builtin_amdgcn_s_wait_asynccnt(0);
#else
    asm volatile("s_wait_asynccnt 0x0" ::: "memory");
#endif
}

// ---------------------------------------------------------------------------
// Kernel 1: per-pixel channel L2 normalize (NCHW f32 -> channels-last f16)
// One wave32 per pixel; lane handles 4 contiguous channels.
// ---------------------------------------------------------------------------
__global__ void __launch_bounds__(256)
normalize_kernel(const float* __restrict__ pred, const float* __restrict__ tgt,
                 _Float16* __restrict__ predN, _Float16* __restrict__ tgtN) {
    const int wave = threadIdx.x >> 5;
    const int lane = threadIdx.x & 31;
    int pix = blockIdx.x * 8 + wave;

    const float* src;
    _Float16* dst;
    int p;
    if (pix < PPIX) { src = pred; dst = predN; p = pix; }
    else            { src = tgt;  dst = tgtN;  p = pix - PPIX; if (p >= TPIX) return; }

    const int b   = p / (HW * HW);
    const int rem = p % (HW * HW);
    const long base = (long)b * CCH * HW * HW + rem;   // + c*HW*HW gives (b,c,y,x)

    float v[4];
    float s = 0.f;
#pragma unroll
    for (int i = 0; i < 4; ++i) {
        int c = lane * 4 + i;
        v[i] = src[base + (long)c * HW * HW];
        s += v[i] * v[i];
    }
#pragma unroll
    for (int m = 16; m >= 1; m >>= 1) s += __shfl_xor(s, m, 32);
    const float inv = 1.0f / fmaxf(sqrtf(s), 1e-12f);

    v4h o;
#pragma unroll
    for (int i = 0; i < 4; ++i) o[i] = (_Float16)(v[i] * inv);
    *(v4h*)(dst + (long)p * CCH + lane * 4) = o;
}

// ---------------------------------------------------------------------------
// Kernel 2: init packed argmax keys
// ---------------------------------------------------------------------------
__global__ void init_best(unsigned long long* __restrict__ best) {
    int i = blockIdx.x * blockDim.x + threadIdx.x;
    if (i < BATCH * NQ) best[i] = 0ull;
}

// ---------------------------------------------------------------------------
// Kernel 3: cosine-score GEMM (implicit 3x3-patch GEMM) with WMMA + argmax.
// Block = 128 threads (4 waves, 2x2 wave grid). Tile: 64 queries x 64 cands;
// each wave owns a 32x32 sub-tile (2 A-frags x 2 B-frags -> 4 WMMAs/step).
// Double-buffered LDS fed by async global->LDS copies; one barrier per step.
// ---------------------------------------------------------------------------
__global__ void __launch_bounds__(128)
match_gemm_kernel(const _Float16* __restrict__ predN,
                  const _Float16* __restrict__ tgtN,
                  unsigned long long* __restrict__ best) {
    __shared__ _Float16 As[2][64 * 32] __attribute__((aligned(16)));  // [buf][m][k]
    __shared__ _Float16 Bs[2][64 * 32] __attribute__((aligned(16)));  // [buf][n][k]

    const int b     = blockIdx.z;
    const int mbase = blockIdx.y * 64;
    const int nbase = blockIdx.x * 64;
    const int tid   = threadIdx.x;
    const int lane  = tid & 31;
    const int wave  = tid >> 5;
    const int lane16 = lane & 15;
    const int hi     = lane >> 4;
    const int wm     = wave >> 1;   // wave row (0..1)
    const int wn     = wave & 1;    // wave col (0..1)

    v8f acc[4];                     // [ms*2+ns]
#pragma unroll
    for (int m = 0; m < 4; ++m)
#pragma unroll
        for (int v = 0; v < 8; ++v) acc[m][v] = 0.0f;

    // Cooperative-load assignment: thread -> (tile row, 16-half chunk pair)
    const int lr = tid >> 1;   // 0..63
    const int lp = tid & 1;    // which 32B half of the 64B row

    int qa  = mbase + lr;  if (qa >= NQ)   qa = NQ - 1;   // clamp; dummies masked later
    int ja  = nbase + lr;  if (ja >= NTOT) ja = NTOT - 1;
    const int qy = qa / NHW, qx = qa % NHW;
    const int kk = ja / NQ,  nn = ja % NQ;
    const int ny = nn / NHW, nx = nn % NHW;

    const unsigned ldsA0 = (unsigned)(unsigned long long)&As[0][lr * 32 + lp * 16];
    const unsigned ldsA1 = (unsigned)(unsigned long long)&As[1][lr * 32 + lp * 16];
    const unsigned ldsB0 = (unsigned)(unsigned long long)&Bs[0][lr * 32 + lp * 16];
    const unsigned ldsB1 = (unsigned)(unsigned long long)&Bs[1][lr * 32 + lp * 16];

    auto issue_tiles = [&](int step, int buf) {
        const int tap = step >> 2;                      // 0..8 -> (di,dj)
        const int c0  = ((step & 3) << 5) + lp * 16;
        const int di  = tap / 3, dj = tap % 3;
        const _Float16* sa = predN +
            (((long)(b * HW + qy + di) * HW + (qx + dj)) * CCH + c0);
        const _Float16* sb = tgtN +
            ((((long)(b * TOPK + kk) * HW + (ny + di)) * HW + (nx + dj)) * CCH + c0);
        const unsigned da = buf ? ldsA1 : ldsA0;
        const unsigned db = buf ? ldsB1 : ldsB0;
        async_cp16(da,      (unsigned long long)sa);
        async_cp16(da + 16, (unsigned long long)(sa + 8));
        async_cp16(db,      (unsigned long long)sb);
        async_cp16(db + 16, (unsigned long long)(sb + 8));
    };

    issue_tiles(0, 0);
    int cur = 0;

    for (int step = 0; step < NSTEP; ++step) {
        wait_async0();          // this wave's copies into buf[cur] landed
        __syncthreads();        // everyone's copies landed; prev reads of other buf done
        if (step + 1 < NSTEP) issue_tiles(step + 1, cur ^ 1);

        const _Float16* Ac = As[cur];
        const _Float16* Bc = Bs[cur];

        // A fragments (16x32): lane group hi=0: K={0..7,16..23}, hi=1:
        // K={8..15,24..31}; M = lane16.
        v16h af[2], bf[2];
#pragma unroll
        for (int ms = 0; ms < 2; ++ms) {
            const _Float16* ar = Ac + (wm * 32 + ms * 16 + lane16) * 32;
            v8h a0 = *(const v8h*)(ar + hi * 8);
            v8h a1 = *(const v8h*)(ar + 16 + hi * 8);
            af[ms] = __builtin_shufflevector(a0, a1,
                0,1,2,3,4,5,6,7,8,9,10,11,12,13,14,15);
        }
        // B fragments (32x16): lanes 0-15 K={0..7,16..23}, lanes 16-31 K={8..15,24..31};
        // N = lane16 -> two 16-byte chunks of the [n][k] row.
#pragma unroll
        for (int ns = 0; ns < 2; ++ns) {
            const _Float16* br = Bc + (wn * 32 + ns * 16 + lane16) * 32;
            v8h b0 = *(const v8h*)(br + hi * 8);
            v8h b1 = *(const v8h*)(br + 16 + hi * 8);
            bf[ns] = __builtin_shufflevector(b0, b1,
                0,1,2,3,4,5,6,7,8,9,10,11,12,13,14,15);
        }
#pragma unroll
        for (int ms = 0; ms < 2; ++ms)
#pragma unroll
            for (int ns = 0; ns < 2; ++ns)
                acc[ms * 2 + ns] = __builtin_amdgcn_wmma_f32_16x16x32_f16(
                    false, af[ms], false, bf[ns], (short)0,
                    acc[ms * 2 + ns], false, false);

        cur ^= 1;
    }

    // Per-query max/argmax: merge the two N-subtiles in registers, reduce the
    // 16 column-lanes, then global combine via packed-u64 atomicMax
    // ("largest score, smallest index" == reference tie-breaking).
    const int rowBase = mbase + wm * 32;
    const int colBase = nbase + wn * 32;
#pragma unroll
    for (int ms = 0; ms < 2; ++ms) {
#pragma unroll
        for (int v = 0; v < 8; ++v) {
            const int c0i = colBase + lane16;
            const int c1i = colBase + 16 + lane16;
            float v0 = (c0i < NTOT) ? acc[ms * 2 + 0][v] : -3.0e38f;
            float v1 = (c1i < NTOT) ? acc[ms * 2 + 1][v] : -3.0e38f;
            float val = v0;  int ci = c0i;                 // tie -> smaller idx
            if (v1 > v0) { val = v1; ci = c1i; }
#pragma unroll
            for (int msk = 8; msk >= 1; msk >>= 1) {
                float ov = __shfl_xor(val, msk, 32);
                int   oc = __shfl_xor(ci,  msk, 32);
                if (ov > val || (ov == val && oc < ci)) { val = ov; ci = oc; }
            }
            if (lane16 == 0) {
                const int q = rowBase + ms * 16 + v + 8 * hi;  // D layout: M=v+8*hi
                if (q < NQ) {
                    unsigned u = __float_as_uint(val);
                    u = (u & 0x80000000u) ? ~u : (u | 0x80000000u);
                    unsigned long long key =
                        ((unsigned long long)u << 32) |
                        (unsigned long long)(0xFFFFFFFFu - (unsigned)ci);
                    atomicMax(best + (long)b * NQ + q, key);
                }
            }
        }
    }
}

// ---------------------------------------------------------------------------
// Kernel 4: gather winning raw target patch, SSE vs raw pred patch.
// One wave32 per (b, q); 36 elements per lane.
// ---------------------------------------------------------------------------
__global__ void __launch_bounds__(256)
gather_mse_kernel(const float* __restrict__ pred, const float* __restrict__ tgt,
                  const unsigned long long* __restrict__ best,
                  float* __restrict__ partial) {
    const int wave = threadIdx.x >> 5;
    const int lane = threadIdx.x & 31;
    const int id = blockIdx.x * 8 + wave;
    if (id >= BATCH * NQ) return;
    const int b = id / NQ, q = id % NQ;

    const unsigned long long key = best[id];
    const unsigned cand = 0xFFFFFFFFu - (unsigned)(key & 0xFFFFFFFFull);
    const int k = cand / NQ, n = cand % NQ;
    const int qy = q / NHW, qx = q % NHW;
    const int ny = n / NHW, nx = n % NHW;

    const float* pbase = pred + (long)b * CCH * HW * HW;
    const float* tbase = tgt + (long)(b * TOPK + k) * CCH * HW * HW;

    float s = 0.f;
#pragma unroll 4
    for (int i = 0; i < KDIM / 32; ++i) {
        const int e = lane + i * 32;
        const int c = e / 9, t = e % 9;
        const int di = t / 3, dj = t % 3;
        const float pv = pbase[((long)c * HW + qy + di) * HW + qx + dj];
        const float tv = tbase[((long)c * HW + ny + di) * HW + nx + dj];
        const float d = pv - tv;
        s += d * d;
    }
#pragma unroll
    for (int m = 16; m >= 1; m >>= 1) s += __shfl_xor(s, m, 32);
    if (lane == 0) partial[id] = s;
}

// ---------------------------------------------------------------------------
// Kernel 5: deterministic final reduction -> mean
// ---------------------------------------------------------------------------
__global__ void __launch_bounds__(256)
final_reduce_kernel(const float* __restrict__ partial, float* __restrict__ out) {
    __shared__ float sm[256];
    float s = 0.f;
    for (int i = threadIdx.x; i < BATCH * NQ; i += 256) s += partial[i];
    sm[threadIdx.x] = s;
    __syncthreads();
    for (int w = 128; w >= 1; w >>= 1) {
        if ((int)threadIdx.x < w) sm[threadIdx.x] += sm[threadIdx.x + w];
        __syncthreads();
    }
    if (threadIdx.x == 0)
        out[0] = sm[0] / (float)((long)BATCH * NQ * KDIM);
}

// ---------------------------------------------------------------------------
extern "C" void kernel_launch(void* const* d_in, const int* in_sizes, int n_in,
                              void* d_out, int out_size, void* d_ws, size_t ws_size,
                              hipStream_t stream) {
    const float* pred = (const float*)d_in[0];   // (2,128,48,48) f32
    const float* tgt  = (const float*)d_in[1];   // (4,128,48,48) f32

    char* ws = (char*)d_ws;
    _Float16* predN = (_Float16*)ws;                           // 1.18 MB
    _Float16* tgtN  = predN + (size_t)PPIX * CCH;              // 2.36 MB
    unsigned long long* best = (unsigned long long*)(tgtN + (size_t)TPIX * CCH);
    float* partial = (float*)(best + BATCH * NQ);
    float* out = (float*)d_out;

    normalize_kernel<<<(PPIX + TPIX) / 8, 256, 0, stream>>>(pred, tgt, predN, tgtN);
    init_best<<<(BATCH * NQ + 255) / 256, 256, 0, stream>>>(best);

    dim3 g2((NTOT + 63) / 64, (NQ + 63) / 64, BATCH);
    match_gemm_kernel<<<g2, 128, 0, stream>>>(predN, tgtN, best);

    gather_mse_kernel<<<(BATCH * NQ + 7) / 8, 256, 0, stream>>>(pred, tgt, best, partial);
    final_reduce_kernel<<<1, 256, 0, stream>>>(partial, out);
}
